// SimAttention_88630945120837
// MI455X (gfx1250) — compile-verified, matrix operand order checked by reading, not run on hardware
//
#include <hip/hip_runtime.h>
#include <hip/hip_bf16.h>

// MI455X / gfx1250, wave32. bf16 WMMA (V_WMMA_F32_16X16X32_BF16) for all GEMMs.
// B operands pre-swizzled into per-lane WMMA fragment layout (L2-resident),
// A operands kept/produced as bf16 row-major so fragments are two aligned
// global b128 loads. No LDS staging in the GEMM hot loops.

typedef __attribute__((ext_vector_type(16))) __bf16 v16bf;
typedef __attribute__((ext_vector_type(8)))  float  v8f;

__device__ __forceinline__ unsigned short f2bf(float f) {
  unsigned u = __float_as_uint(f);
  u += 0x7FFFu + ((u >> 16) & 1u);          // RNE to bf16
  return (unsigned short)(u >> 16);
}
__device__ __forceinline__ float bf2f(unsigned short s) {
  return __uint_as_float(((unsigned)s) << 16);
}

// 16-bit A fragment per CDNA5 ISA layout: lane = half*16 + lr, row = lr.
// VGPR v holds K-pair k = (v<4 ? 2v : 2v+8) + 8*half  ->  for a fixed half the
// 16 bf16 are two contiguous 16B runs: dwords [4*half..4*half+3] (k 0..15 side)
// and [8+4*half..8+4*half+3] (k 16..31 side).
__device__ __forceinline__ v16bf load_a_frag(const unsigned short* rowptr, int half) {
  union { v16bf v; uint4 q[2]; } r;
  const uint4* p = (const uint4*)rowptr;    // rowptr 16B-aligned (K%8==0, kk%32==0)
  r.q[0] = p[half];
  r.q[1] = p[2 + half];
  return r.v;
}

// B fragment: pre-swizzled buffer stores, per (ktile,ntile), 32 lanes x 16 bf16
// contiguous -> 2 coalesced b128 loads per lane.
__device__ __forceinline__ v16bf load_b_frag(const unsigned short* tilebase, int lane) {
  union { v16bf v; uint4 q[2]; } r;
  const uint4* p = (const uint4*)(tilebase + lane * 16);
  r.q[0] = p[0];
  r.q[1] = p[1];
  return r.v;
}

// ---------------------------------------------------------------------------
// Pre-swizzle weight matrix B[K,N] (fp32) into WMMA B-fragment order (bf16):
// out[((kt*ntiles + nt)*32 + lane)*16 + e] = bf16(B[kt*32 + k(e,lane)][nt*16 + lr])
// ---------------------------------------------------------------------------
__global__ __launch_bounds__(256)
void swizzleB(const float* __restrict__ B, unsigned short* __restrict__ out,
              int K, int N)
{
  int idx = blockIdx.x * 256 + threadIdx.x;
  if (idx >= K * N) return;
  int e    = idx & 15;
  int lane = (idx >> 4) & 31;
  int tile = idx >> 9;
  int ntiles = N >> 4;
  int kt = tile / ntiles, nt = tile % ntiles;
  int half = lane >> 4, lr = lane & 15;
  int vv = e >> 1;
  int k = ((vv < 4) ? 2 * vv : 2 * vv + 8) + 8 * half + (e & 1);
  out[idx] = f2bf(B[(size_t)(kt * 32 + k) * N + nt * 16 + lr]);
}

// fp32 -> bf16 bulk convert (8 elems/thread, b128 in / b128 out)
__global__ __launch_bounds__(256)
void cvt_bf16(const float* __restrict__ in, unsigned short* __restrict__ out, int n8)
{
  int i = blockIdx.x * 256 + threadIdx.x;
  if (i >= n8) return;
  float4 f0 = ((const float4*)in)[2 * i];
  float4 f1 = ((const float4*)in)[2 * i + 1];
  union { unsigned short s[8]; uint4 q; } pk;
  pk.s[0] = f2bf(f0.x); pk.s[1] = f2bf(f0.y); pk.s[2] = f2bf(f0.z); pk.s[3] = f2bf(f0.w);
  pk.s[4] = f2bf(f1.x); pk.s[5] = f2bf(f1.y); pk.s[6] = f2bf(f1.z); pk.s[7] = f2bf(f1.w);
  ((uint4*)out)[i] = pk.q;
}

// ---------------------------------------------------------------------------
// GEMM: C[M,N] = act(A[M,K](bf16) @ Bsw(K,N swizzled bf16) + bias)
// Block = 128 thr (4 waves). Wave owns 32x64 tile (2x4 wmma accs). No LDS.
// flags: bit0 = relu, bit1 = output bf16 (else fp32).
// Requires M%128==0, N%64==0, K%32==0.
// ---------------------------------------------------------------------------
__global__ __launch_bounds__(128)
void gemm_bf16(const unsigned short* __restrict__ A, const unsigned short* __restrict__ Bsw,
               const float* __restrict__ bias, void* __restrict__ Cout,
               int M, int N, int K, int flags)
{
  const int tid  = threadIdx.x;
  const int wave = tid >> 5, lane = tid & 31;
  const int half = lane >> 4, lr = lane & 15;
  const int m0 = blockIdx.y * 128 + wave * 32;
  const int n0 = blockIdx.x * 64;
  const int ntiles = N >> 4;

  v8f acc[2][4];
  #pragma unroll
  for (int mi = 0; mi < 2; ++mi)
    #pragma unroll
    for (int nt = 0; nt < 4; ++nt)
      #pragma unroll
      for (int j = 0; j < 8; ++j) acc[mi][nt][j] = 0.f;

  const unsigned short* ap0 = A + (size_t)(m0 + lr) * K;
  const unsigned short* ap1 = A + (size_t)(m0 + 16 + lr) * K;

  for (int kk = 0; kk < K; kk += 32) {
    v16bf a0 = load_a_frag(ap0 + kk, half);
    v16bf a1 = load_a_frag(ap1 + kk, half);
    const unsigned short* bbase =
        Bsw + ((size_t)(kk >> 5) * ntiles + (n0 >> 4)) * 512;
    #pragma unroll
    for (int nt = 0; nt < 4; ++nt) {
      v16bf bf = load_b_frag(bbase + nt * 512, lane);
      acc[0][nt] = __builtin_amdgcn_wmma_f32_16x16x32_bf16(
          false, a0, false, bf, (short)0, acc[0][nt], false, false);
      acc[1][nt] = __builtin_amdgcn_wmma_f32_16x16x32_bf16(
          false, a1, false, bf, (short)0, acc[1][nt], false, false);
    }
  }

  #pragma unroll
  for (int mi = 0; mi < 2; ++mi)
    #pragma unroll
    for (int nt = 0; nt < 4; ++nt) {
      int n = n0 + nt * 16 + lr;
      float bv = bias ? bias[n] : 0.f;
      #pragma unroll
      for (int r = 0; r < 8; ++r) {
        int m = m0 + mi * 16 + r + 8 * half;     // C/D: VGPR r -> M = r + 8*half
        float v = acc[mi][nt][r] + bv;
        if (flags & 1) v = fmaxf(v, 0.f);
        if (flags & 2) ((unsigned short*)Cout)[(size_t)m * N + n] = f2bf(v);
        else           ((float*)Cout)[(size_t)m * N + n] = v;
      }
    }
}

// ---------------------------------------------------------------------------
// GEMM2: q2[b,s,:] = relu( sum_kh gather(q, perm[b,kh]) @ K2[kh] + b2 )
// K=1024 with A rows gathered per 256-wide kh band; K2 pre-swizzled as [1024,256].
// ---------------------------------------------------------------------------
__global__ __launch_bounds__(128)
void gemm2_bf16(const unsigned short* __restrict__ qbf, const int* __restrict__ perm,
                const unsigned short* __restrict__ Bsw, const float* __restrict__ b2,
                unsigned short* __restrict__ q2bf)
{
  const int S = 4096, DL = 256, KH = 4;
  const int tid  = threadIdx.x;
  const int wave = tid >> 5, lane = tid & 31;
  const int half = lane >> 4, lr = lane & 15;
  const int b  = blockIdx.z;
  const int m0 = blockIdx.y * 128 + wave * 32;
  const int n0 = blockIdx.x * 64;
  const int ntiles = DL >> 4;   // 16

  v8f acc[2][4];
  #pragma unroll
  for (int mi = 0; mi < 2; ++mi)
    #pragma unroll
    for (int nt = 0; nt < 4; ++nt)
      #pragma unroll
      for (int j = 0; j < 8; ++j) acc[mi][nt][j] = 0.f;

  for (int kk = 0; kk < KH * DL; kk += 32) {
    int kh = kk >> 8, d0 = kk & 255;
    int src0 = perm[(size_t)(b * KH + kh) * S + m0 + lr];
    int src1 = perm[(size_t)(b * KH + kh) * S + m0 + 16 + lr];
    v16bf a0 = load_a_frag(qbf + (size_t)(b * S + src0) * DL + d0, half);
    v16bf a1 = load_a_frag(qbf + (size_t)(b * S + src1) * DL + d0, half);
    const unsigned short* bbase =
        Bsw + ((size_t)(kk >> 5) * ntiles + (n0 >> 4)) * 512;
    #pragma unroll
    for (int nt = 0; nt < 4; ++nt) {
      v16bf bf = load_b_frag(bbase + nt * 512, lane);
      acc[0][nt] = __builtin_amdgcn_wmma_f32_16x16x32_bf16(
          false, a0, false, bf, (short)0, acc[0][nt], false, false);
      acc[1][nt] = __builtin_amdgcn_wmma_f32_16x16x32_bf16(
          false, a1, false, bf, (short)0, acc[1][nt], false, false);
    }
  }

  #pragma unroll
  for (int mi = 0; mi < 2; ++mi)
    #pragma unroll
    for (int nt = 0; nt < 4; ++nt) {
      int n = n0 + nt * 16 + lr;
      float bv = b2[n];
      #pragma unroll
      for (int r = 0; r < 8; ++r) {
        int m = m0 + mi * 16 + r + 8 * half;
        q2bf[((size_t)b * S + m) * DL + n] = f2bf(fmaxf(acc[mi][nt][r] + bv, 0.f));
      }
    }
}

// ---------------------------------------------------------------------------
// Column variance of RMVs over S per (b,k): one block per (b,k).
// ---------------------------------------------------------------------------
__global__ __launch_bounds__(256)
void colstats(const float* __restrict__ rmv, float* __restrict__ hvars)
{
  const int S = 4096, K = 64;
  int b = blockIdx.x >> 6, k = blockIdx.x & 63;
  __shared__ float s1[256], s2[256];
  float a = 0.f, a2 = 0.f;
  for (int i = threadIdx.x; i < S; i += 256) {
    float v = rmv[((size_t)b * S + i) * K + k];
    a += v; a2 += v * v;
  }
  s1[threadIdx.x] = a; s2[threadIdx.x] = a2;
  __syncthreads();
  for (int off = 128; off > 0; off >>= 1) {
    if (threadIdx.x < off) { s1[threadIdx.x] += s1[threadIdx.x + off];
                             s2[threadIdx.x] += s2[threadIdx.x + off]; }
    __syncthreads();
  }
  if (threadIdx.x == 0) {
    float mu = s1[0] / (float)S;
    hvars[b * K + k] = s2[0] / (float)S - mu * mu;
  }
}

// Top-4 columns by variance, stable descending (matches argsort(-h_vars)).
__global__ void top4_select(const float* __restrict__ hvars, int* __restrict__ top)
{
  int b = blockIdx.x, k = threadIdx.x;   // 64 threads
  float v = hvars[b * 64 + k];
  int rank = 0;
  for (int j = 0; j < 64; ++j) {
    float u = hvars[b * 64 + j];
    rank += (u > v) || (u == v && j < k);
  }
  if (rank < 4) top[b * 4 + rank] = k;
}

// Stable ascending argsort of h_max[b,kh,:] via O(S^2) rank count + scatter.
__global__ __launch_bounds__(256)
void rank_perm(const float* __restrict__ rmv, const int* __restrict__ top,
               int* __restrict__ perm)
{
  const int S = 4096, K = 64, KH = 4;
  int b = blockIdx.z, kh = blockIdx.y;
  int s = blockIdx.x * 256 + threadIdx.x;
  int k = top[b * 4 + kh];
  float v = rmv[((size_t)b * S + s) * K + k];
  __shared__ float buf[256];
  int rank = 0;
  for (int j0 = 0; j0 < S; j0 += 256) {
    __syncthreads();
    buf[threadIdx.x] = rmv[((size_t)b * S + j0 + threadIdx.x) * K + k];
    __syncthreads();
    #pragma unroll 8
    for (int t = 0; t < 256; ++t) {
      float u = buf[t];
      int j = j0 + t;
      rank += (u < v) || (u == v && j < s);
    }
  }
  perm[((size_t)b * KH + kh) * S + rank] = s;
}

// ---------------------------------------------------------------------------
// Fused: window softmax(q2.q2_nb/16) -> out = sum_w p_w*u[row_w] -> residual
// -> LayerNorm. q2/u read as bf16. One block (256 thr) per (b,s), 4 h/thread.
// ---------------------------------------------------------------------------
__global__ __launch_bounds__(256)
void attn_ln(const unsigned short* __restrict__ q2b, const unsigned short* __restrict__ ub,
             const float* __restrict__ inp, const float* __restrict__ gamma,
             const float* __restrict__ beta, float* __restrict__ out)
{
  const int S = 4096, DL = 256, DH = 1024;
  int bs = blockIdx.x;
  int b = bs >> 12, s = bs & 4095;
  int t = threadIdx.x;

  __shared__ float qrow[256];
  __shared__ float red[256];
  __shared__ float dots[12];

  int rows[9];
  #pragma unroll
  for (int w = 0; w < 9; ++w) {
    int r = (s < 4) ? w : (s >= S - 4) ? (S - 9 + w) : (s - 4 + w);
    rows[w] = b * S + r;
  }

  qrow[t] = bf2f(q2b[(size_t)bs * DL + t]);
  __syncthreads();

  for (int w = 0; w < 9; ++w) {
    red[t] = qrow[t] * bf2f(q2b[(size_t)rows[w] * DL + t]);
    __syncthreads();
    for (int off = 128; off > 0; off >>= 1) {
      if (t < off) red[t] += red[t + off];
      __syncthreads();
    }
    if (t == 0) dots[w] = red[0];
    __syncthreads();
  }

  float mx = -1e30f;
  #pragma unroll
  for (int w = 0; w < 9; ++w) mx = fmaxf(mx, dots[w] * 0.0625f);
  float p[9], psum = 0.f;
  #pragma unroll
  for (int w = 0; w < 9; ++w) { p[w] = __expf(dots[w] * 0.0625f - mx); psum += p[w]; }
  float pinv = 1.f / psum;
  #pragma unroll
  for (int w = 0; w < 9; ++w) p[w] *= pinv;

  float x[4] = {0.f, 0.f, 0.f, 0.f};
  #pragma unroll
  for (int w = 0; w < 9; ++w) {
    const unsigned short* ur = ub + (size_t)rows[w] * DH;
    float pw = p[w];
    #pragma unroll
    for (int j = 0; j < 4; ++j) x[j] += pw * bf2f(ur[t + j * 256]);
  }
  #pragma unroll
  for (int j = 0; j < 4; ++j) x[j] += inp[(size_t)bs * DH + t + j * 256];

  float ps  = x[0] + x[1] + x[2] + x[3];
  float ps2 = x[0]*x[0] + x[1]*x[1] + x[2]*x[2] + x[3]*x[3];
  red[t] = ps;  __syncthreads();
  for (int off = 128; off > 0; off >>= 1) {
    if (t < off) red[t] += red[t + off];
    __syncthreads();
  }
  float mu = red[0] / (float)DH;
  __syncthreads();
  red[t] = ps2; __syncthreads();
  for (int off = 128; off > 0; off >>= 1) {
    if (t < off) red[t] += red[t + off];
    __syncthreads();
  }
  float var = red[0] / (float)DH - mu * mu;
  float rinv = rsqrtf(var + 0.001f);
  #pragma unroll
  for (int j = 0; j < 4; ++j) {
    int h = t + j * 256;
    out[(size_t)bs * DH + h] = gamma[h] * (x[j] - mu) * rinv + beta[h];
  }
}

// ---------------------------------------------------------------------------
extern "C" void kernel_launch(void* const* d_in, const int* in_sizes, int n_in,
                              void* d_out, int out_size, void* d_ws, size_t ws_size,
                              hipStream_t stream)
{
  const int B = 4, S = 4096, DH = 1024, DL = 256, K = 64, KH = 4;
  const int M = B * S;   // 16384

  const float* inputs = (const float*)d_in[0];
  const float* W1     = (const float*)d_in[1];
  const float* b1     = (const float*)d_in[2];
  const float* RME    = (const float*)d_in[3];
  const float* K2     = (const float*)d_in[4];
  const float* b2     = (const float*)d_in[5];
  const float* Wu     = (const float*)d_in[6];
  const float* bu     = (const float*)d_in[7];
  const float* gamma  = (const float*)d_in[8];
  const float* beta   = (const float*)d_in[9];
  float* out = (float*)d_out;

  // workspace carve-up (all regions 16B-aligned), ~90 MB total
  char* ws = (char*)d_ws;
  unsigned short* Xbf  = (unsigned short*)ws;  ws += (size_t)M * DH * 2;   // 33.5MB
  unsigned short* qbf  = (unsigned short*)ws;  ws += (size_t)M * DL * 2;   // 8.4MB
  unsigned short* q2bf = (unsigned short*)ws;  ws += (size_t)M * DL * 2;   // 8.4MB
  unsigned short* ubf  = (unsigned short*)ws;  ws += (size_t)M * DH * 2;   // 33.5MB
  float* rmv = (float*)ws;                     ws += (size_t)M * K * 4;    // 4.2MB
  float* hv  = (float*)ws;                     ws += B * K * 4;
  int* top   = (int*)ws;                       ws += B * KH * 4;
  int* perm  = (int*)ws;                       ws += (size_t)B * KH * S * 4;
  unsigned short* swW1  = (unsigned short*)ws; ws += (size_t)DH * DL * 2;
  unsigned short* swRME = (unsigned short*)ws; ws += (size_t)DL * K * 2;
  unsigned short* swK2  = (unsigned short*)ws; ws += (size_t)KH * DL * DL * 2;
  unsigned short* swWu  = (unsigned short*)ws;

  // 0) one-time bf16 conversion / weight swizzles (weights stay L2-resident)
  cvt_bf16<<<(M * DH / 8 + 255) / 256, 256, 0, stream>>>(inputs, Xbf, M * DH / 8);
  swizzleB<<<(DH * DL + 255) / 256, 256, 0, stream>>>(W1, swW1, DH, DL);
  swizzleB<<<(DL * K + 255) / 256, 256, 0, stream>>>(RME, swRME, DL, K);
  swizzleB<<<(KH * DL * DL + 255) / 256, 256, 0, stream>>>(K2, swK2, KH * DL, DL);
  swizzleB<<<(DL * DH + 255) / 256, 256, 0, stream>>>(Wu, swWu, DL, DH);

  // 1) q = relu(inputs @ W1 + b1) -> bf16
  gemm_bf16<<<dim3(DL / 64, M / 128), 128, 0, stream>>>(Xbf, swW1, b1, qbf, M, DL, DH, 3);

  // 2) RMVs = q @ RME -> fp32
  gemm_bf16<<<dim3(K / 64, M / 128), 128, 0, stream>>>(qbf, swRME, nullptr, rmv, M, K, DL, 0);

  // 3) column variances, top-4, stable argsort permutation
  colstats<<<B * K, 256, 0, stream>>>(rmv, hv);
  top4_select<<<B, K, 0, stream>>>(hv, top);
  rank_perm<<<dim3(S / 256, KH, B), 256, 0, stream>>>(rmv, top, perm);

  // 4) q2 = relu(sum_kh gather(q) @ K2 + b2) -> bf16
  gemm2_bf16<<<dim3(DL / 64, S / 128, B), 128, 0, stream>>>(qbf, perm, swK2, b2, q2bf);

  // 5) u = relu(q2 @ Wu + bu) -> bf16 (val rows computed ONCE per row, not x9)
  gemm_bf16<<<dim3(DH / 64, M / 128), 128, 0, stream>>>(q2bf, swWu, bu, ubf, M, DH, DL, 3);

  // 6) window softmax + weighted-u + residual + LayerNorm
  attn_ln<<<M, 256, 0, stream>>>(q2bf, ubf, inputs, gamma, beta, out);
}
